// DIN_8753143349927
// MI455X (gfx1250) — matrix operand order repaired and checked by reading, not compiled, not run on hardware
//
#include <hip/hip_runtime.h>

typedef __bf16 bf16_t;
typedef __attribute__((ext_vector_type(16))) __bf16 v16bf;
typedef __attribute__((ext_vector_type(8)))  __bf16 v8bf;
typedef __attribute__((ext_vector_type(8)))  float  v8f;

#define B_SZ 4096
#define L_SZ 200
#define D_SZ 32

__device__ __forceinline__ bf16_t f2bf(float f) {
  // round-to-nearest-even fp32 -> bf16
  unsigned u = __float_as_uint(f);
  unsigned r = (u + 0x7FFFu + ((u >> 16) & 1u)) >> 16;
  return __builtin_bit_cast(bf16_t, (unsigned short)r);
}

// Swap bits 3<->4 of a 0..31 index: maps within-chunk element index d to its
// position in the 16-bit WMMA A-matrix lane layout (low-half lanes read
// positions 0..15 contiguously = K {0..7,16..23}; high-half read 16..31).
__device__ __forceinline__ int swap34(int d) {
  return (d & 7) | ((d & 16) >> 1) | ((d & 8) << 1);
}

// 32-byte bf16 fragment load from 16-byte-aligned LDS/global memory.
__device__ __forceinline__ v16bf load16(const bf16_t* p) {
  union { v16bf v; v8bf h[2]; } u;
  u.h[0] = *(const v8bf*)(p);
  u.h[1] = *(const v8bf*)(p + 8);
  return u.v;
}

__device__ __forceinline__ v16bf bfsub(v16bf a, v16bf b) { return a - b; }
__device__ __forceinline__ v16bf bfmul(v16bf a, v16bf b) { return a * b; }

// ---------------------------------------------------------------------------
// Kernel 0: convert weight matrices to bf16 (w1 zero-padded K: 489 -> 512)
// ---------------------------------------------------------------------------
__global__ __launch_bounds__(256)
void convert_weights(const float* __restrict__ w1, const float* __restrict__ w2,
                     const float* __restrict__ wa1,
                     bf16_t* __restrict__ w1b, bf16_t* __restrict__ w2b,
                     bf16_t* __restrict__ wa1b) {
  int i = blockIdx.x * 256 + threadIdx.x;
  if (i < 512 * 128) {
    int k = i >> 7;
    w1b[(i & 127) * 512 + (k & ~31) + swap34(k & 31)] = (k < 489) ? f2bf(w1[i]) : f2bf(0.0f);
  }
  if (i < 128 * 64) { int k = i >> 6; w2b[(i & 63) * 128 + (k & ~31) + swap34(k & 31)] = f2bf(w2[i]); }
  if (i < 128 * 32) { int k = i >> 5; wa1b[(i & 31) * 128 + (k & ~31) + swap34(k & 31)] = f2bf(wa1[i]); }
}

// ---------------------------------------------------------------------------
// Kernel 1: DIN attention. grid (B, 2): y==0 -> cate, y==1 -> brand.
// 256 threads = 8 waves; each wave owns M-tiles of 16 sequence rows.
// Score GEMM: [208,128] @ [128,32] via v_wmma_f32_16x16x32_bf16.
// K-chunk 0 (broadcast target) is hoisted: Cz = te x W computed once.
// ---------------------------------------------------------------------------
#define HA_STRIDE 40   // 80B rows: 16-byte aligned, off-power-of-2 banking

__global__ __launch_bounds__(256)
void din_attention(const float* __restrict__ emb_cate, const int* __restrict__ idx_cate,
                   const int* __restrict__ hist_cate,
                   const float* __restrict__ emb_brand, const int* __restrict__ idx_brand,
                   const int* __restrict__ hist_brand,
                   const int* __restrict__ seq_len,
                   const bf16_t* __restrict__ wa1b,   // [32][128] bf16, n-major, k A-swizzled
                   const float* __restrict__ b_a1, const float* __restrict__ w_a2,
                   const float* __restrict__ b_a2,
                   float* __restrict__ i_cate, float* __restrict__ i_brand) {
  __shared__ float h_s[L_SZ][D_SZ];                    // fp32 history (weighted sum)
  __shared__ __align__(16) bf16_t hA[208 * HA_STRIDE]; // bf16 history, A-swizzled, 0-padded
  __shared__ __align__(16) bf16_t tA[32];              // bf16 target, A-swizzled
  __shared__ float scores[208];
  __shared__ float red[256];
  __shared__ float wsum[8][D_SZ];

  const int b = blockIdx.x;
  const bool is_brand = (blockIdx.y != 0);
  const float* emb  = is_brand ? emb_brand  : emb_cate;
  const int*   idx  = is_brand ? idx_brand  : idx_cate;
  const int*   hist = is_brand ? hist_brand : hist_cate;
  float*       outp = is_brand ? i_brand    : i_cate;

  const int tid  = threadIdx.x;
  const int lane = tid & 31;
  const int wave = tid >> 5;
  const bool hi  = lane >= 16;

  // Target embedding: fp32 kept implicitly via tA; bf16 swizzled copy for WMMA.
  if (tid < D_SZ) tA[swap34(tid)] = f2bf(emb[(long)idx[b] * D_SZ + tid]);
  // History gather: fp32 rows (l<200) + bf16 A-swizzled rows (208, zero-padded).
  for (int i = tid; i < 208 * D_SZ; i += 256) {
    int l = i >> 5, d = i & 31;
    float v = 0.0f;
    if (l < L_SZ) {
      v = emb[(long)hist[b * L_SZ + l] * D_SZ + d];
      h_s[l][d] = v;
    }
    hA[l * HA_STRIDE + swap34(d)] = f2bf(v);
  }
  __syncthreads();

  // B fragments of w_a1 (K=128 -> 4 chunks of 32; N=32 -> 2 tiles), per-wave regs.
  const int kk = (lane & 15) + (hi ? 16 : 0);
  v16bf Bf[4][2];
#pragma unroll
  for (int c = 0; c < 4; ++c)
#pragma unroll
    for (int nt = 0; nt < 2; ++nt)
      Bf[c][nt] = load16(wa1b + (16 * nt + (lane & 15)) * 128 + 32 * c + (hi ? 16 : 0));

  // Hoisted chunk 0: A rows are all the broadcast target te.
  const v16bf tf = load16(&tA[hi ? 16 : 0]);
  v8f Czero = {};
  const v8f Cz0 = __builtin_amdgcn_wmma_f32_16x16x32_bf16(false, tf, false, Bf[0][0],
                                                          (short)0, Czero, false, false);
  const v8f Cz1 = __builtin_amdgcn_wmma_f32_16x16x32_bf16(false, tf, false, Bf[0][1],
                                                          (short)0, Czero, false, false);

  const float ba2   = b_a2[0];
  const int   ncol  = lane & 15;
  const float bias0 = b_a1[ncol],      wa20 = w_a2[ncol];
  const float bias1 = b_a1[16 + ncol], wa21 = w_a2[16 + ncol];

  for (int mt = wave; mt < 13; mt += 8) {       // 13 tiles cover 208 rows
    const int l0 = mt * 16;
    const int l  = l0 + (lane & 15);
    // Chunk 1: h ; chunk 2: te - h ; chunk 3: te * h (bf16 packed math, no LDS)
    const v16bf hf = load16(&hA[l * HA_STRIDE + (hi ? 16 : 0)]);
    const v16bf df = bfsub(tf, hf);
    const v16bf pf = bfmul(tf, hf);
    v8f C0 = Cz0, C1 = Cz1;
    C0 = __builtin_amdgcn_wmma_f32_16x16x32_bf16(false, hf, false, Bf[1][0], (short)0, C0, false, false);
    C1 = __builtin_amdgcn_wmma_f32_16x16x32_bf16(false, hf, false, Bf[1][1], (short)0, C1, false, false);
    C0 = __builtin_amdgcn_wmma_f32_16x16x32_bf16(false, df, false, Bf[2][0], (short)0, C0, false, false);
    C1 = __builtin_amdgcn_wmma_f32_16x16x32_bf16(false, df, false, Bf[2][1], (short)0, C1, false, false);
    C0 = __builtin_amdgcn_wmma_f32_16x16x32_bf16(false, pf, false, Bf[3][0], (short)0, C0, false, false);
    C1 = __builtin_amdgcn_wmma_f32_16x16x32_bf16(false, pf, false, Bf[3][1], (short)0, C1, false, false);

    // score[m] = sum_n relu(C[m,n]+b)*w_a2[n]; n = lane&15, m = v + 8*hi
#pragma unroll
    for (int v = 0; v < 8; ++v) {
      float x = fmaxf(C0[v] + bias0, 0.0f) * wa20
              + fmaxf(C1[v] + bias1, 0.0f) * wa21;
#pragma unroll
      for (int msk = 1; msk < 16; msk <<= 1) x += __shfl_xor(x, msk, 32);
      if ((lane & 15) == 0) scores[l0 + v + (hi ? 8 : 0)] = x + ba2;
    }
  }
  __syncthreads();

  // Masked softmax over L
  const int slen = seq_len[b];
  const bool valid = (tid < L_SZ) && (tid < slen);
  float sc = valid ? scores[tid] : -1e30f;
  red[tid] = sc;
  __syncthreads();
  for (int s = 128; s > 0; s >>= 1) {
    if (tid < s) red[tid] = fmaxf(red[tid], red[tid + s]);
    __syncthreads();
  }
  float mx = red[0];
  __syncthreads();
  float e = valid ? __expf(sc - mx) : 0.0f;
  red[tid] = e;
  __syncthreads();
  for (int s = 128; s > 0; s >>= 1) {
    if (tid < s) red[tid] += red[tid + s];
    __syncthreads();
  }
  float inv = 1.0f / red[0];
  __syncthreads();
  if (tid < L_SZ) scores[tid] = e * inv;
  __syncthreads();

  // Weighted sum over history (fp32 h in LDS)
  {
    int d = tid & 31, ch = tid >> 5;
    float acc = 0.0f;
    for (int l = ch; l < L_SZ; l += 8) acc += scores[l] * h_s[l][d];
    wsum[ch][d] = acc;
  }
  __syncthreads();
  if (tid < D_SZ) {
    float s = 0.0f;
#pragma unroll
    for (int c = 0; c < 8; ++c) s += wsum[c][tid];
    outp[b * D_SZ + tid] = s;
  }
}

// ---------------------------------------------------------------------------
// Kernel 2: fused embedding gather + 3-layer DNN. 128 threads = 4 waves,
// block handles 16 batch rows. GEMM1 [16,512]@[512,128], GEMM2 [16,128]@[128,64]
// via WMMA (A from swizzled LDS, vector loads); GEMM3 via shuffle reduction.
// ---------------------------------------------------------------------------
#define XS_STRIDE 520   // 1040B rows: 16B aligned, banks spread
#define H1_STRIDE 136   // 272B rows

struct SparseArgs {
  const float* emb[11];
  const int*   idx[11];
};

__global__ __launch_bounds__(128)
void din_dnn(SparseArgs sp,
             const float* __restrict__ emb_cate, const int* __restrict__ idx_cate,
             const float* __restrict__ emb_brand, const int* __restrict__ idx_brand,
             const float* __restrict__ i_cate, const float* __restrict__ i_brand,
             const float* __restrict__ dense,
             const bf16_t* __restrict__ w1b, const float* __restrict__ b1,
             const bf16_t* __restrict__ w2b, const float* __restrict__ b2,
             const float* __restrict__ w3, const float* __restrict__ b3,
             float* __restrict__ out) {
  __shared__ __align__(16) bf16_t Xs[16 * XS_STRIDE];
  __shared__ __align__(16) bf16_t h1s[16 * H1_STRIDE];
  __shared__ float owave[4][16];

  const int tid  = threadIdx.x;
  const int lane = tid & 31;
  const int wave = tid >> 5;
  const bool hi  = lane >= 16;
  const int b0   = blockIdx.x * 16;

  // Build dnn_input tile [16 x 489] (padded to 512), stored A-swizzled bf16.
  for (int i = tid; i < 16 * 512; i += 128) {
    int r = i >> 9, col = i & 511;
    int b = b0 + r;
    float val;
    if (col < 352) {                      // 11 sparse features x 32
      int f = col >> 5, d = col & 31;
      val = sp.emb[f][(long)sp.idx[f][b] * 32 + d];
    } else if (col < 384) {               // t_cate
      val = emb_cate[(long)idx_cate[b] * 32 + (col - 352)];
    } else if (col < 416) {               // t_brand
      val = emb_brand[(long)idx_brand[b] * 32 + (col - 384)];
    } else if (col < 448) {               // i_cate
      val = i_cate[b * 32 + (col - 416)];
    } else if (col < 480) {               // i_brand
      val = i_brand[b * 32 + (col - 448)];
    } else if (col < 489) {               // dense
      val = dense[b * 9 + (col - 480)];
    } else {
      val = 0.0f;                         // K padding
    }
    Xs[r * XS_STRIDE + (col & ~31) + swap34(col & 31)] = f2bf(val);
  }
  __syncthreads();

  const int kk  = (lane & 15) + (hi ? 16 : 0);
  const int m_a = lane & 15;
  const int aoff = hi ? 16 : 0;

  // GEMM1: wave w owns output columns 32w..32w+31 (two 16-wide n-tiles)
  {
    v8f C0 = {}, C1 = {};
    const int nb = 32 * wave;
#pragma unroll 4
    for (int c = 0; c < 16; ++c) {
      v16bf A   = load16(&Xs[m_a * XS_STRIDE + 32 * c + aoff]);
      v16bf Bf0 = load16(w1b + (nb + (lane & 15)) * 512 + 32 * c + aoff);
      v16bf Bf1 = load16(w1b + (nb + 16 + (lane & 15)) * 512 + 32 * c + aoff);
      C0 = __builtin_amdgcn_wmma_f32_16x16x32_bf16(false, A, false, Bf0,
                                                   (short)0, C0, false, false);
      C1 = __builtin_amdgcn_wmma_f32_16x16x32_bf16(false, A, false, Bf1,
                                                   (short)0, C1, false, false);
    }
    const int n_0 = nb + (lane & 15);
    const int n_1 = nb + 16 + (lane & 15);
    const int p_0 = (n_0 & ~31) + swap34(n_0 & 31);   // store A-swizzled for GEMM2
    const int p_1 = (n_1 & ~31) + swap34(n_1 & 31);
    const float bb0 = b1[n_0], bb1 = b1[n_1];
#pragma unroll
    for (int v = 0; v < 8; ++v) {
      int m = v + (hi ? 8 : 0);
      h1s[m * H1_STRIDE + p_0] = f2bf(fmaxf(C0[v] + bb0, 0.0f));
      h1s[m * H1_STRIDE + p_1] = f2bf(fmaxf(C1[v] + bb1, 0.0f));
    }
  }
  __syncthreads();

  // GEMM2: wave w owns output columns 16w..16w+15; then fold w3 (64->1)
  {
    v8f C2 = {};
#pragma unroll
    for (int c = 0; c < 4; ++c) {
      v16bf A  = load16(&h1s[m_a * H1_STRIDE + 32 * c + aoff]);
      v16bf Bf = load16(w2b + (16 * wave + (lane & 15)) * 128 + 32 * c + aoff);
      C2 = __builtin_amdgcn_wmma_f32_16x16x32_bf16(false, A, false, Bf,
                                                   (short)0, C2, false, false);
    }
    const int n = 16 * wave + (lane & 15);
    const float bb = b2[n], w3v = w3[n];
#pragma unroll
    for (int v = 0; v < 8; ++v) {
      float x = fmaxf(C2[v] + bb, 0.0f) * w3v;
#pragma unroll
      for (int msk = 1; msk < 16; msk <<= 1) x += __shfl_xor(x, msk, 32);
      if ((lane & 15) == 0) owave[wave][v + (hi ? 8 : 0)] = x;
    }
  }
  __syncthreads();
  if (tid < 16)
    out[b0 + tid] = owave[0][tid] + owave[1][tid] + owave[2][tid] + owave[3][tid] + b3[0];
}

// ---------------------------------------------------------------------------
extern "C" void kernel_launch(void* const* d_in, const int* in_sizes, int n_in,
                              void* d_out, int out_size, void* d_ws, size_t ws_size,
                              hipStream_t stream) {
  (void)in_sizes; (void)n_in; (void)out_size; (void)ws_size;

  const float* emb_adgroup = (const float*)d_in[0];  const int* idx_adgroup = (const int*)d_in[1];
  const float* emb_pid     = (const float*)d_in[2];  const int* idx_pid     = (const int*)d_in[3];
  const float* emb_cate    = (const float*)d_in[4];  const int* idx_cate    = (const int*)d_in[5];
  const float* emb_brand   = (const float*)d_in[6];  const int* idx_brand   = (const int*)d_in[7];
  const float* emb_cmsg    = (const float*)d_in[8];  const int* idx_cmsg    = (const int*)d_in[9];
  const float* emb_cmss    = (const float*)d_in[10]; const int* idx_cmss    = (const int*)d_in[11];
  const float* emb_gender  = (const float*)d_in[12]; const int* idx_gender  = (const int*)d_in[13];
  const float* emb_occ     = (const float*)d_in[14]; const int* idx_occ     = (const int*)d_in[15];
  const float* emb_age     = (const float*)d_in[16]; const int* idx_age     = (const int*)d_in[17];
  const float* emb_shop    = (const float*)d_in[18]; const int* idx_shop    = (const int*)d_in[19];
  const float* emb_gcc     = (const float*)d_in[20]; const int* idx_gcc     = (const int*)d_in[21];
  const float* emb_hour    = (const float*)d_in[22]; const int* idx_hour    = (const int*)d_in[23];
  const float* emb_wday    = (const float*)d_in[24]; const int* idx_wday    = (const int*)d_in[25];
  const int*   hist_cate   = (const int*)d_in[26];
  const int*   hist_brand  = (const int*)d_in[27];
  const int*   seq_len     = (const int*)d_in[28];
  const float* dense       = (const float*)d_in[29];
  const float* w_a1 = (const float*)d_in[30]; const float* b_a1 = (const float*)d_in[31];
  const float* w_a2 = (const float*)d_in[32]; const float* b_a2 = (const float*)d_in[33];
  const float* w1   = (const float*)d_in[34]; const float* b1   = (const float*)d_in[35];
  const float* w2   = (const float*)d_in[36]; const float* b2   = (const float*)d_in[37];
  const float* w3   = (const float*)d_in[38]; const float* b3   = (const float*)d_in[39];

  // Workspace layout
  char*   ws      = (char*)d_ws;
  float*  i_cate  = (float*)(ws);                         // 4096*32*4  = 524288
  float*  i_brand = (float*)(ws + 524288);                // 524288
  bf16_t* w1b     = (bf16_t*)(ws + 1048576);              // 512*128*2  = 131072
  bf16_t* w2b     = (bf16_t*)(ws + 1179648);              // 128*64*2   = 16384
  bf16_t* wa1b    = (bf16_t*)(ws + 1196032);              // 128*32*2   = 8192

  convert_weights<<<dim3(256), dim3(256), 0, stream>>>(w1, w2, w_a1, w1b, w2b, wa1b);

  din_attention<<<dim3(B_SZ, 2), dim3(256), 0, stream>>>(
      emb_cate, idx_cate, hist_cate, emb_brand, idx_brand, hist_brand,
      seq_len, wa1b, b_a1, w_a2, b_a2, i_cate, i_brand);

  SparseArgs sp;
  sp.emb[0] = emb_adgroup; sp.idx[0] = idx_adgroup;
  sp.emb[1] = emb_pid;     sp.idx[1] = idx_pid;
  sp.emb[2] = emb_cmsg;    sp.idx[2] = idx_cmsg;
  sp.emb[3] = emb_cmss;    sp.idx[3] = idx_cmss;
  sp.emb[4] = emb_gender;  sp.idx[4] = idx_gender;
  sp.emb[5] = emb_occ;     sp.idx[5] = idx_occ;
  sp.emb[6] = emb_age;     sp.idx[6] = idx_age;
  sp.emb[7] = emb_shop;    sp.idx[7] = idx_shop;
  sp.emb[8] = emb_gcc;     sp.idx[8] = idx_gcc;
  sp.emb[9] = emb_hour;    sp.idx[9] = idx_hour;
  sp.emb[10] = emb_wday;   sp.idx[10] = idx_wday;

  din_dnn<<<dim3(B_SZ / 16), dim3(128), 0, stream>>>(
      sp, emb_cate, idx_cate, emb_brand, idx_brand, i_cate, i_brand, dense,
      w1b, b1, w2b, b2, w3, b3, (float*)d_out);
}
